// nakliTransformerDecoder_23381801960275
// MI455X (gfx1250) — compile-verified, matrix operand order checked by reading, not run on hardware
//
#include <hip/hip_runtime.h>
#include <hip/hip_bf16.h>
#include <math.h>

// ---------------------------------------------------------------------------
// Transformer decoder block for MI455X (gfx1250, wave32, WMMA).
// fp32 I/O, f16 WMMA operands, f32 accumulation everywhere.
// Round 3: fix async-builtin pointer types (int4* in AS1 / AS3).
// ---------------------------------------------------------------------------

typedef _Float16 v8h  __attribute__((ext_vector_type(8)));
typedef _Float16 v16h __attribute__((ext_vector_type(16)));
typedef float    v8f  __attribute__((ext_vector_type(8)));
typedef int      v4i  __attribute__((ext_vector_type(4)));

constexpr int Bc  = 4;
constexpr int Sc  = 2048;
constexpr int Dc  = 1024;
constexpr int Hc  = 16;
constexpr int HDc = 64;
constexpr int DFFc = 4096;
constexpr int Mc  = Bc * Sc;   // 8192 rows

// ---- async global->LDS (CDNA5), with synchronous fallback ------------------

#if defined(__has_builtin)
#if __has_builtin(__builtin_amdgcn_global_load_async_to_lds_b128) && \
    __has_builtin(__builtin_amdgcn_s_wait_asynccnt)
#define USE_ASYNC_LDS 1
#endif
#endif

typedef __attribute__((address_space(1))) v4i* as1_v4i_ptr;
typedef __attribute__((address_space(3))) v4i* as3_v4i_ptr;

__device__ __forceinline__ void async_cp16(const _Float16* g, _Float16* l) {
#ifdef USE_ASYNC_LDS
  __builtin_amdgcn_global_load_async_to_lds_b128(
      (as1_v4i_ptr)g, (as3_v4i_ptr)l, 0, 0);
#else
  *(v8h*)l = *(const v8h*)g;
#endif
}

template <int N>
__device__ __forceinline__ void async_wait() {
#ifdef USE_ASYNC_LDS
  __builtin_amdgcn_s_wait_asynccnt(N);
#endif
}

// ---- WMMA helpers ----------------------------------------------------------

__device__ __forceinline__ v8f wmma_f16(v16h a, v16h b, v8f c) {
  return __builtin_amdgcn_wmma_f32_16x16x32_f16(
      /*neg_a=*/false, a, /*neg_b=*/false, b,
      /*c_mod=*/(short)0, c, /*reuse_a=*/false, /*reuse_b=*/false);
}

// Load a 16x32 f16 A-style fragment from an LDS tile with row stride `ldk`
// (halves). Lanes 0-15: rows 0..15, K 0..7 & 16..23; lanes 16-31: K 8..15 &
// 24..31. Serves B too when the B tile is stored N-major ("Bt[N][K]").
__device__ __forceinline__ v16h ldfragA(const _Float16* base, int ldk) {
  unsigned lane = threadIdx.x & 31u;
  const _Float16* p = base + (lane & 15u) * ldk + ((lane >> 4) << 3);
  v8h lo = *(const v8h*)(p);
  v8h hi = *(const v8h*)(p + 16);
  return __builtin_shufflevector(lo, hi,
      0, 1, 2, 3, 4, 5, 6, 7, 8, 9, 10, 11, 12, 13, 14, 15);
}

// ---- weight convert + transpose: W[K][N] f32 -> Wt[N][K] f16 ---------------

__global__ void k_transpose(const float* __restrict__ W, _Float16* __restrict__ Wt,
                            int K, int N) {
  size_t idx = (size_t)blockIdx.x * blockDim.x + threadIdx.x;
  if (idx >= (size_t)K * N) return;
  int k = (int)(idx / N);
  int n = (int)(idx % N);
  Wt[(size_t)n * K + k] = (_Float16)W[idx];
}

// ---- layernorm: f32 row -> f16 normalized row ------------------------------

__global__ void __launch_bounds__(256)
k_layernorm(const float* __restrict__ x, const float* __restrict__ g,
            const float* __restrict__ b, _Float16* __restrict__ out) {
  __shared__ float red[2][8];
  const int row = blockIdx.x;
  const float* xr = x + (size_t)row * Dc;
  float v[4];
  float s = 0.f, ss = 0.f;
#pragma unroll
  for (int i = 0; i < 4; ++i) {
    v[i] = xr[threadIdx.x + i * 256];
    s += v[i];
    ss += v[i] * v[i];
  }
#pragma unroll
  for (int m = 16; m >= 1; m >>= 1) {
    s  += __shfl_xor(s, m);
    ss += __shfl_xor(ss, m);
  }
  const int w = threadIdx.x >> 5;
  if ((threadIdx.x & 31) == 0) { red[0][w] = s; red[1][w] = ss; }
  __syncthreads();
  s = 0.f; ss = 0.f;
#pragma unroll
  for (int i = 0; i < 8; ++i) { s += red[0][i]; ss += red[1][i]; }
  const float mu   = s * (1.0f / Dc);
  const float var  = ss * (1.0f / Dc) - mu * mu;
  const float rstd = rsqrtf(var + 1e-5f);
#pragma unroll
  for (int i = 0; i < 4; ++i) {
    const int c = threadIdx.x + i * 256;
    out[(size_t)row * Dc + c] = (_Float16)((v[i] - mu) * rstd * g[c] + b[c]);
  }
}

// ---- generic WMMA GEMM: C[M,N] = A[M,K] * Bt[N,K]^T ------------------------
// 128 threads (4 waves), block tile 128x128, wave tile 64x64, BK=64 halves,
// double-buffered async global->LDS staging (16 ASYNCcnt ops per wave/stage).
// EPI 0: outH = (f16)acc                (QKV projections)
// EPI 1: outF = res + acc + bias        (residual adds)
// EPI 2: outH = (f16)gelu(acc + bias)   (FFN up-proj)

constexpr int BKh = 64;   // K halves staged per buffer
constexpr int LDT = 72;   // padded LDS row stride (halves) -> conflict-free

template <int EPI>
__global__ void __launch_bounds__(128)
k_gemm(const _Float16* __restrict__ A, const _Float16* __restrict__ Bt,
       int M, int N, int K,
       const float* __restrict__ bias, const float* __restrict__ res,
       float* __restrict__ outF, _Float16* __restrict__ outH) {
  __shared__ __align__(16) _Float16 sA[2][128 * LDT];
  __shared__ __align__(16) _Float16 sB[2][128 * LDT];
  const int m0 = blockIdx.y * 128;
  const int n0 = blockIdx.x * 128;
  const int t  = threadIdx.x;
  const int w  = t >> 5;
  const int wm = (w >> 1) * 64;
  const int wn = (w & 1) * 64;

  v8f acc[4][4];
  const v8f vzero = {0.f, 0.f, 0.f, 0.f, 0.f, 0.f, 0.f, 0.f};
#pragma unroll
  for (int i = 0; i < 4; ++i)
#pragma unroll
    for (int j = 0; j < 4; ++j) acc[i][j] = vzero;

  const int nT = K / BKh;
  const _Float16* ga0 = A  + (size_t)(m0 + t) * K;
  const _Float16* gb0 = Bt + (size_t)(n0 + t) * K;

  auto stage = [&](int kt, int buf) {
    const _Float16* ga = ga0 + kt * BKh;
    const _Float16* gb = gb0 + kt * BKh;
    _Float16* la = &sA[buf][t * LDT];
    _Float16* lb = &sB[buf][t * LDT];
#pragma unroll
    for (int c = 0; c < BKh; c += 8) async_cp16(ga + c, la + c);
#pragma unroll
    for (int c = 0; c < BKh; c += 8) async_cp16(gb + c, lb + c);
  };

  stage(0, 0);
  for (int kt = 0; kt < nT; ++kt) {
    const int buf = kt & 1;
    if (kt + 1 < nT) {
      stage(kt + 1, buf ^ 1);
      async_wait<16>();   // previous stage's 16 async ops complete (in-order)
    } else {
      async_wait<0>();
    }
    __syncthreads();

#pragma unroll
    for (int ks = 0; ks < BKh; ks += 32) {
      v16h af[4], bf[4];
#pragma unroll
      for (int i = 0; i < 4; ++i)
        af[i] = ldfragA(&sA[buf][(wm + i * 16) * LDT + ks], LDT);
#pragma unroll
      for (int i = 0; i < 4; ++i)
        bf[i] = ldfragA(&sB[buf][(wn + i * 16) * LDT + ks], LDT);
#pragma unroll
      for (int i = 0; i < 4; ++i)
#pragma unroll
        for (int j = 0; j < 4; ++j) acc[i][j] = wmma_f16(af[i], bf[j], acc[i][j]);
    }
    __syncthreads();  // all waves done with buf before it is restaged
  }

  // epilogue: lane holds column (lane&15), rows rbase..rbase+7 per fragment
  const unsigned lane = t & 31u;
  const int rbase = (int)((lane >> 4) << 3);
  const int cl = (int)(lane & 15u);
#pragma unroll
  for (int i = 0; i < 4; ++i) {
#pragma unroll
    for (int j = 0; j < 4; ++j) {
      const int colg = n0 + wn + j * 16 + cl;
      const float bc = (EPI != 0 && bias != nullptr) ? bias[colg] : 0.0f;
#pragma unroll
      for (int r = 0; r < 8; ++r) {
        const int rowg = m0 + wm + i * 16 + rbase + r;
        const size_t idx = (size_t)rowg * N + colg;
        float v = acc[i][j][r];
        if (EPI == 0) {
          outH[idx] = (_Float16)v;
        } else if (EPI == 1) {
          outF[idx] = res[idx] + v + bc;
        } else {
          v += bc;
          const float ge = 0.5f * v * (1.0f + erff(v * 0.70710678118654752f));
          outH[idx] = (_Float16)ge;
        }
      }
    }
  }
}

// ---- flash attention over packed qkv f16 [M][3*D] --------------------------
// Block = (q-tile of 64 rows, head, batch); 128 threads = 4 waves, 16 q rows
// per wave. Online softmax, never materializes the SxS score matrix.

__global__ void __launch_bounds__(128)
k_flash(const _Float16* __restrict__ qkv, _Float16* __restrict__ ctx) {
  __shared__ __align__(16) _Float16 Qs[64 * 72];
  __shared__ __align__(16) _Float16 Ks[64 * 72];
  __shared__ __align__(16) _Float16 Vt[64 * 72];
  __shared__ __align__(16) _Float16 Ps[4][16 * 72];

  const int qt   = blockIdx.x;   // 0..31
  const int head = blockIdx.y;   // 0..15
  const int b    = blockIdx.z;   // 0..3
  const int t    = threadIdx.x;
  const int w    = t >> 5;
  const unsigned lane = t & 31u;
  const size_t rowstride = 3 * Dc;
  const size_t base = (size_t)b * Sc * rowstride + (size_t)head * HDc;
  const int r  = t >> 1;
  const int c0 = (t & 1) * 32;

  // stage Q (pre-scaled by 1/sqrt(HD) = 0.125)
  {
    const _Float16* src = qkv + base + (size_t)(qt * 64 + r) * rowstride + c0;
    _Float16* dst = &Qs[r * 72 + c0];
#pragma unroll
    for (int c = 0; c < 32; ++c) dst[c] = (_Float16)((float)src[c] * 0.125f);
  }
  __syncthreads();
  const v16h qf0 = ldfragA(&Qs[(w * 16) * 72 + 0], 72);
  const v16h qf1 = ldfragA(&Qs[(w * 16) * 72 + 32], 72);

  const v8f vzero = {0.f, 0.f, 0.f, 0.f, 0.f, 0.f, 0.f, 0.f};
  v8f o[4];
#pragma unroll
  for (int n = 0; n < 4; ++n) o[n] = vzero;
  float m_[8], l_[8];
#pragma unroll
  for (int rr = 0; rr < 8; ++rr) { m_[rr] = -3.0e38f; l_[rr] = 0.0f; }

  for (int j = 0; j <= qt; ++j) {
    __syncthreads();  // protect Ks/Vt from previous iteration's readers
    {
      const size_t kvrow = base + (size_t)(j * 64 + r) * rowstride;
      // K tile: async copy, 4x b128 per thread
      const _Float16* ksrc = qkv + kvrow + Dc + c0;
      _Float16* kdst = &Ks[r * 72 + c0];
#pragma unroll
      for (int c = 0; c < 32; c += 8) async_cp16(ksrc + c, kdst + c);
      // V tile: transposed store (synchronous)
      const _Float16* vsrc = qkv + kvrow + 2 * Dc + c0;
#pragma unroll
      for (int c = 0; c < 32; ++c) Vt[(c0 + c) * 72 + r] = vsrc[c];
      // prefetch next KV tile into cache while this one is consumed
      if (j < qt) {
        const _Float16* nsrc =
            qkv + base + (size_t)((j + 1) * 64 + r) * rowstride + Dc + c0;
        __builtin_prefetch(nsrc, 0, 0);
        __builtin_prefetch(nsrc + Dc, 0, 0);
      }
      async_wait<0>();
    }
    __syncthreads();

    // scores S = Q * K^T  (Ks rows are kv index = B's N dimension)
    v8f s[4];
#pragma unroll
    for (int ni = 0; ni < 4; ++ni) {
      s[ni] = vzero;
      const v16h kf0 = ldfragA(&Ks[(ni * 16) * 72 + 0], 72);
      const v16h kf1 = ldfragA(&Ks[(ni * 16) * 72 + 32], 72);
      s[ni] = wmma_f16(qf0, kf0, s[ni]);
      s[ni] = wmma_f16(qf1, kf1, s[ni]);
    }

    if (j == qt) {  // causal mask on diagonal tile only
#pragma unroll
      for (int ni = 0; ni < 4; ++ni)
#pragma unroll
        for (int rr = 0; rr < 8; ++rr) {
          const int qrow = w * 16 + (int)((lane >> 4) << 3) + rr;  // local in 64
          const int kcol = ni * 16 + (int)(lane & 15u);
          if (kcol > qrow) s[ni][rr] = -3.0e38f;
        }
    }

    // online softmax per row (row owned by 16 lanes of one half-wave)
    float al[8];
#pragma unroll
    for (int rr = 0; rr < 8; ++rr) {
      float lm = fmaxf(fmaxf(s[0][rr], s[1][rr]), fmaxf(s[2][rr], s[3][rr]));
#pragma unroll
      for (int msk = 8; msk >= 1; msk >>= 1) lm = fmaxf(lm, __shfl_xor(lm, msk));
      const float nm = fmaxf(m_[rr], lm);
      al[rr] = __expf(m_[rr] - nm);
      m_[rr] = nm;
      float psum = 0.f;
#pragma unroll
      for (int ni = 0; ni < 4; ++ni) {
        const float p = __expf(s[ni][rr] - nm);
        s[ni][rr] = p;
        psum += p;
      }
#pragma unroll
      for (int msk = 8; msk >= 1; msk >>= 1) psum += __shfl_xor(psum, msk);
      l_[rr] = l_[rr] * al[rr] + psum;
    }

    // rescale running O, spill P through per-wave LDS to A-fragment layout
#pragma unroll
    for (int ni = 0; ni < 4; ++ni)
#pragma unroll
      for (int rr = 0; rr < 8; ++rr) {
        o[ni][rr] *= al[rr];
        const int rl = (int)((lane >> 4) << 3) + rr;
        Ps[w][rl * 72 + ni * 16 + (int)(lane & 15u)] = (_Float16)s[ni][rr];
      }
    const v16h pf0 = ldfragA(&Ps[w][0], 72);
    const v16h pf1 = ldfragA(&Ps[w][32], 72);
#pragma unroll
    for (int n = 0; n < 4; ++n) {
      const v16h vf0 = ldfragA(&Vt[(n * 16) * 72 + 0], 72);
      const v16h vf1 = ldfragA(&Vt[(n * 16) * 72 + 32], 72);
      o[n] = wmma_f16(pf0, vf0, o[n]);
      o[n] = wmma_f16(pf1, vf1, o[n]);
    }
  }

  // normalize and store ctx as f16 [b, s, head*64 + hd]
#pragma unroll
  for (int rr = 0; rr < 8; ++rr) {
    const float inv = 1.0f / l_[rr];
    const int qrow = qt * 64 + w * 16 + (int)((lane >> 4) << 3) + rr;
#pragma unroll
    for (int n = 0; n < 4; ++n) {
      const int hd = n * 16 + (int)(lane & 15u);
      ctx[(size_t)(b * Sc + qrow) * Dc + head * HDc + hd] =
          (_Float16)(o[n][rr] * inv);
    }
  }
}

// ---------------------------------------------------------------------------

extern "C" void kernel_launch(void* const* d_in, const int* in_sizes, int n_in,
                              void* d_out, int out_size, void* d_ws, size_t ws_size,
                              hipStream_t stream) {
  const float* x    = (const float*)d_in[0];
  const float* ln1g = (const float*)d_in[1];
  const float* ln1b = (const float*)d_in[2];
  const float* Wq   = (const float*)d_in[3];
  const float* Wk   = (const float*)d_in[4];
  const float* Wv   = (const float*)d_in[5];
  const float* Wo   = (const float*)d_in[6];
  const float* ln2g = (const float*)d_in[7];
  const float* ln2b = (const float*)d_in[8];
  const float* W1   = (const float*)d_in[9];
  const float* b1   = (const float*)d_in[10];
  const float* W2   = (const float*)d_in[11];
  const float* b2   = (const float*)d_in[12];
  float* out = (float*)d_out;

  char* ws = (char*)d_ws;
  size_t off = 0;
  auto alloc = [&](size_t bytes) -> char* {
    char* p = ws + off;
    off += (bytes + 255) & ~(size_t)255;
    return p;
  };
  _Float16* WqkvT = (_Float16*)alloc((size_t)3 * Dc * Dc * 2);   //  6 MB
  _Float16* WoT   = (_Float16*)alloc((size_t)Dc * Dc * 2);       //  2 MB
  _Float16* W1T   = (_Float16*)alloc((size_t)DFFc * Dc * 2);     //  8 MB
  _Float16* W2T   = (_Float16*)alloc((size_t)Dc * DFFc * 2);     //  8 MB
  _Float16* h16   = (_Float16*)alloc((size_t)Mc * Dc * 2);       // 16 MB (reused as h2)
  _Float16* qkv   = (_Float16*)alloc((size_t)Mc * 3 * Dc * 2);   // 48 MB
  _Float16* ctxb  = (_Float16*)alloc((size_t)Mc * Dc * 2);       // 16 MB
  float*    x1    = (float*)   alloc((size_t)Mc * Dc * 4);       // 32 MB
  // FFN intermediate (64 MB) reuses the dead qkv+ctx span exactly.
  _Float16* hff   = qkv;

  const int T = 256;
  auto blocks = [&](size_t n) { return (unsigned)((n + T - 1) / T); };

  // weights -> f16, transposed to [N][K]
  k_transpose<<<blocks((size_t)Dc * Dc), T, 0, stream>>>(Wq, WqkvT, Dc, Dc);
  k_transpose<<<blocks((size_t)Dc * Dc), T, 0, stream>>>(Wk, WqkvT + (size_t)Dc * Dc, Dc, Dc);
  k_transpose<<<blocks((size_t)Dc * Dc), T, 0, stream>>>(Wv, WqkvT + (size_t)2 * Dc * Dc, Dc, Dc);
  k_transpose<<<blocks((size_t)Dc * Dc), T, 0, stream>>>(Wo, WoT, Dc, Dc);
  k_transpose<<<blocks((size_t)Dc * DFFc), T, 0, stream>>>(W1, W1T, Dc, DFFc);
  k_transpose<<<blocks((size_t)DFFc * Dc), T, 0, stream>>>(W2, W2T, DFFc, Dc);

  // h = LN1(x)  (f16)
  k_layernorm<<<Mc, 256, 0, stream>>>(x, ln1g, ln1b, h16);

  // qkv = h @ [Wq|Wk|Wv]  (f16 out)
  k_gemm<0><<<dim3(3 * Dc / 128, Mc / 128), 128, 0, stream>>>(
      h16, WqkvT, Mc, 3 * Dc, Dc, nullptr, nullptr, nullptr, qkv);

  // ctx = causal_softmax(q k^T / 8) v   (flash, f16 out)
  k_flash<<<dim3(Sc / 64, Hc, Bc), 128, 0, stream>>>(qkv, ctxb);

  // x1 = x + ctx @ Wo   (f32)
  k_gemm<1><<<dim3(Dc / 128, Mc / 128), 128, 0, stream>>>(
      ctxb, WoT, Mc, Dc, Dc, nullptr, x, x1, nullptr);

  // h2 = LN2(x1)  (f16, reuses h16)
  k_layernorm<<<Mc, 256, 0, stream>>>(x1, ln2g, ln2b, h16);

  // hff = gelu(h2 @ W1 + b1)  (f16)
  k_gemm<2><<<dim3(DFFc / 128, Mc / 128), 128, 0, stream>>>(
      h16, W1T, Mc, DFFc, Dc, b1, nullptr, nullptr, hff);

  // out = x1 + hff @ W2 + b2  (f32)
  k_gemm<1><<<dim3(Dc / 128, Mc / 128), 128, 0, stream>>>(
      hff, W2T, Mc, Dc, DFFc, b2, x1, out, nullptr);
}
